// GMMNavigator_80625126081175
// MI455X (gfx1250) — compile-verified, matrix operand order checked by reading, not run on hardware
//
#include <hip/hip_runtime.h>
#include <hip/hip_bf16.h>
#include <math.h>

// ---------------- constants from the reference ----------------
constexpr int T    = 2048;
constexpr int FEAT = 256;
constexpr int HID  = 512;
constexpr int PHI  = 1024;
constexpr int RANK = 8;
constexpr int CMAX = 8;
constexpr int MLP  = 256;
constexpr float DT_MIN = 0.3f, DT_MAX = 3.0f;
constexpr float G_MIN  = 0.1f, G_MAX  = 5.0f;
constexpr float U_MAX  = 2.0f;
constexpr float GS_EPS = 1e-8f;
constexpr float LN_EPS = 1e-5f;

typedef _Float16 v16h __attribute__((ext_vector_type(16)));
typedef _Float16 h8   __attribute__((ext_vector_type(8)));
typedef float    v8f  __attribute__((ext_vector_type(8)));

// ---------------- small helpers ----------------
__device__ __forceinline__ float waveReduceSum(float v) {
  #pragma unroll
  for (int o = 16; o > 0; o >>= 1) v += __shfl_down(v, o, 32);
  return v;
}
__device__ __forceinline__ float softplusf(float x) {
  return (x > 20.f) ? x : log1pf(expf(x));
}
__device__ __forceinline__ float sigmoidf(float x) {
  return 1.f / (1.f + expf(-x));
}
__device__ __forceinline__ float gelu_exact(float x) {
  return 0.5f * x * (1.f + erff(x * 0.70710678118654752f));
}

// ---- CDNA5 async global->LDS copy (16 B per lane, tracked by ASYNCcnt) ----
__device__ __forceinline__ void async_copy16(_Float16* lds, const _Float16* g) {
  // flat LDS-aperture address truncates to the LDS byte offset (ISA 10.2)
  unsigned off = (unsigned)(unsigned long long)lds;
  asm volatile("global_load_async_to_lds_b128 %0, %1, off"
               :: "v"(off), "v"(g) : "memory");
}
__device__ __forceinline__ void wait_async0() {
  asm volatile("s_wait_asynccnt 0x0" ::: "memory");
}

// ---------------- init: zero h buffer + barrier state ----------------
__global__ void k_init(float* hbuf, unsigned* sync) {
  int i = threadIdx.x;
  if (i < HID) hbuf[i] = 0.f;
  if (i < 2)   sync[i] = 0u;
}

// ---------------- f32 -> f16 conversion ----------------
__global__ void k_cvt16(const float* __restrict__ in, _Float16* __restrict__ outp, int n) {
  int i = blockIdx.x * 256 + threadIdx.x;
  if (i < n) outp[i] = (_Float16)in[i];
}

// ---------------- feature LayerNorm -> f16 (one wave per row, 256 cols) ----
__global__ void k_feat_ln(const float* __restrict__ x, const float* __restrict__ g,
                          const float* __restrict__ b, _Float16* __restrict__ outp) {
  int row  = blockIdx.x * 8 + (threadIdx.x >> 5);
  int lane = threadIdx.x & 31;
  const float* xr = x + (size_t)row * FEAT;
  float v[8], s = 0.f, s2 = 0.f;
  #pragma unroll
  for (int i = 0; i < 8; ++i) { v[i] = xr[lane * 8 + i]; s += v[i]; s2 += v[i] * v[i]; }
  s  = waveReduceSum(s);  s  = __shfl(s, 0, 32);
  s2 = waveReduceSum(s2); s2 = __shfl(s2, 0, 32);
  float mean = s / FEAT;
  float var  = s2 / FEAT - mean * mean;
  float rstd = rsqrtf(var + LN_EPS);
  #pragma unroll
  for (int i = 0; i < 8; ++i) {
    int c = lane * 8 + i;
    outp[(size_t)row * FEAT + c] = (_Float16)((v[i] - mean) * rstd * g[c] + b[c]);
  }
}

// ---------------- head LayerNorm: 5 (g,b) pairs, h (T x 512) -> f16 x5 ------
__global__ void k_head_ln(const float* __restrict__ hsrc,
                          const float* g0, const float* b0, const float* g1, const float* b1,
                          const float* g2, const float* b2, const float* g3, const float* b3,
                          const float* g4, const float* b4,
                          _Float16* __restrict__ outp) {
  int row  = blockIdx.x * 8 + (threadIdx.x >> 5);
  int lane = threadIdx.x & 31;
  const float* xr = hsrc + (size_t)row * HID;
  float v[16], s = 0.f, s2 = 0.f;
  #pragma unroll
  for (int i = 0; i < 16; ++i) { v[i] = xr[lane * 16 + i]; s += v[i]; s2 += v[i] * v[i]; }
  s  = waveReduceSum(s);  s  = __shfl(s, 0, 32);
  s2 = waveReduceSum(s2); s2 = __shfl(s2, 0, 32);
  float mean = s / HID;
  float rstd = rsqrtf(s2 / HID - mean * mean + LN_EPS);
  const float* gs[5] = {g0, g1, g2, g3, g4};
  const float* bs[5] = {b0, b1, b2, b3, b4};
  #pragma unroll
  for (int h = 0; h < 5; ++h) {
    #pragma unroll
    for (int i = 0; i < 16; ++i) {
      int c = lane * 16 + i;
      outp[(size_t)h * T * HID + (size_t)row * HID + c] =
          (_Float16)((v[i] - mean) * rstd * gs[h][c] + bs[h][c]);
    }
  }
}

// ---------------- WMMA f16 GEMM, LDS-staged + async double buffering --------
// C = A(MxK) * W^T  (W is NxK row-major).  Block: 128(M) x 64(N), 8 waves,
// each wave owns a 32x32 sub-block (4 wmma accumulators).
// A/B K-tiles (128x32 / 64x32 f16) staged in LDS via global_load_async_to_lds,
// rows padded to 40 halves (80 B) to spread the 64 LDS banks.
// EPI: 0 = bias, f32 out ; 1 = bias + exact GELU, f16 out ;
//      2 = bias + softplus + 0.1, f32 out
template<int EPI>
__global__ __launch_bounds__(256)
void k_gemm16(const _Float16* __restrict__ A,
              const _Float16* __restrict__ W,
              const float*    __restrict__ bias,
              float*          __restrict__ Cf,
              _Float16*       __restrict__ Ch,
              int K, int ldc) {
  __shared__ _Float16 Asm[2][128 * 40];   // 2 x 10 KB
  __shared__ _Float16 Bsm[2][64 * 40];    // 2 x  5 KB
  const int tid  = threadIdx.x;
  const int lane = tid & 31;
  const int w    = tid >> 5;            // 0..7
  const int wy   = w & 3;               // 32-row strip within 128
  const int wx   = w >> 2;              // 32-col strip within 64
  const int m0   = blockIdx.y * 128;
  const int n0   = blockIdx.x * 64;
  const int rc   = lane & 15;
  const int koff = (lane >> 4) * 8;

  const int prow = tid >> 2;            // 0..63
  const int pseg = tid & 3;             // 16B segment within a 64B row

  auto prefetch = [&](int kb, int buf) {
    // A tile: 128 rows x 32 halves -> 512 x 16B chunks (2 per thread)
    async_copy16(&Asm[buf][prow * 40 + pseg * 8],
                 A + (size_t)(m0 + prow) * K + kb + pseg * 8);
    async_copy16(&Asm[buf][(prow + 64) * 40 + pseg * 8],
                 A + (size_t)(m0 + prow + 64) * K + kb + pseg * 8);
    // B tile: 64 rows x 32 halves -> 256 x 16B chunks (1 per thread)
    async_copy16(&Bsm[buf][prow * 40 + pseg * 8],
                 W + (size_t)(n0 + prow) * K + kb + pseg * 8);
  };

  v8f acc[2][2] = {};
  const int nk = K / 32;
  prefetch(0, 0);
  for (int i = 0; i < nk; ++i) {
    const int cur = i & 1;
    wait_async0();
    __syncthreads();                       // tile[cur] visible to all waves
    if (i + 1 < nk) prefetch((i + 1) * 32, cur ^ 1);
    v16h a[2], b[2];
    #pragma unroll
    for (int f = 0; f < 2; ++f) {
      const _Float16* ap = &Asm[cur][(wy * 32 + f * 16 + rc) * 40 + koff];
      h8 alo = *(const h8*)ap, ahi = *(const h8*)(ap + 16);
      const _Float16* bp = &Bsm[cur][(wx * 32 + f * 16 + rc) * 40 + koff];
      h8 blo = *(const h8*)bp, bhi = *(const h8*)(bp + 16);
      #pragma unroll
      for (int j = 0; j < 8; ++j) {
        a[f][j] = alo[j]; a[f][8 + j] = ahi[j];
        b[f][j] = blo[j]; b[f][8 + j] = bhi[j];
      }
    }
    #pragma unroll
    for (int fi = 0; fi < 2; ++fi)
      #pragma unroll
      for (int fj = 0; fj < 2; ++fj)
        acc[fi][fj] = __builtin_amdgcn_wmma_f32_16x16x32_f16(
            false, a[fi], false, b[fj], (short)0, acc[fi][fj], false, false);
  }

  const int mAdd = (lane >= 16) ? 8 : 0;
  #pragma unroll
  for (int fi = 0; fi < 2; ++fi) {
    #pragma unroll
    for (int fj = 0; fj < 2; ++fj) {
      const float bv = bias[n0 + wx * 32 + fj * 16 + rc];
      #pragma unroll
      for (int r = 0; r < 8; ++r) {
        float v = acc[fi][fj][r] + bv;
        size_t idx = (size_t)(m0 + wy * 32 + fi * 16 + r + mAdd) * ldc
                   + (n0 + wx * 32 + fj * 16 + rc);
        if (EPI == 0)      Cf[idx] = v;
        else if (EPI == 1) Ch[idx] = (_Float16)gelu_exact(v);
        else               Cf[idx] = softplusf(v) + 0.1f;
      }
    }
  }
}

// ---------------- GRU scan: 32 WGs, LDS-resident f16 weight slices ----------
#define SCAN_NWG 32
__global__ void k_gru_scan(const float* __restrict__ gi,    // T x 1536
                           const float* __restrict__ Whh,   // 1536 x 512
                           const float* __restrict__ bhh,   // 1536
                           float*       __restrict__ hsout, // T x 512
                           float*       __restrict__ hbuf,  // 512
                           unsigned*    __restrict__ syncp) // [0]=count [1]=gen
{
  __shared__ _Float16 Wl[HID * 48];   // 48 KB: 48 rows, transposed, f16
  __shared__ float    hl[HID];
  __shared__ float    accb[48];
  __shared__ float    bh[48];
  const int g   = blockIdx.x;
  const int tid = threadIdx.x;        // blockDim = 64
  const int j0  = g * 16;             // 16 hidden units per WG
  for (int idx = tid; idx < 48 * HID; idx += 64) {
    int c = idx >> 9, k = idx & (HID - 1);
    int row = (c < 16) ? (j0 + c) : (c < 32) ? (HID + j0 + c - 16) : (2 * HID + j0 + c - 32);
    Wl[k * 48 + c] = (_Float16)Whh[(size_t)row * HID + k];
  }
  if (tid < 48) {
    int c = tid;
    int row = (c < 16) ? (j0 + c) : (c < 32) ? (HID + j0 + c - 16) : (2 * HID + j0 + c - 32);
    bh[c] = bhh[row];
  }
  for (int k = tid; k < HID; k += 64) hl[k] = 0.f;
  __syncthreads();

  for (int t = 0; t < T; ++t) {
    if (tid < 48) {
      float acc = 0.f;
      #pragma unroll 8
      for (int k = 0; k < HID; ++k) acc += hl[k] * (float)Wl[k * 48 + tid];
      accb[tid] = acc;
    }
    __syncthreads();
    if (tid < 16) {
      int j = j0 + tid;
      float hr = accb[tid]      + bh[tid];
      float hz = accb[16 + tid] + bh[16 + tid];
      float hn = accb[32 + tid] + bh[32 + tid];
      const float* git = gi + (size_t)t * 3 * HID;
      float r = sigmoidf(git[j] + hr);
      float z = sigmoidf(git[HID + j] + hz);
      float n = tanhf(git[2 * HID + j] + r * hn);
      float hnew = (1.f - z) * n + z * hl[j];
      hbuf[j] = hnew;
      hsout[(size_t)t * HID + j] = hnew;
    }
    __syncthreads();
    if (tid == 0) {
      __threadfence();
      unsigned gen0 = __hip_atomic_load(syncp + 1, __ATOMIC_ACQUIRE, __HIP_MEMORY_SCOPE_AGENT);
      unsigned a = __hip_atomic_fetch_add(syncp, 1u, __ATOMIC_ACQ_REL, __HIP_MEMORY_SCOPE_AGENT);
      if (a == SCAN_NWG - 1) {
        __hip_atomic_store(syncp, 0u, __ATOMIC_RELAXED, __HIP_MEMORY_SCOPE_AGENT);
        __hip_atomic_fetch_add(syncp + 1, 1u, __ATOMIC_RELEASE, __HIP_MEMORY_SCOPE_AGENT);
      } else {
        while (__hip_atomic_load(syncp + 1, __ATOMIC_ACQUIRE, __HIP_MEMORY_SCOPE_AGENT) == gen0)
          __builtin_amdgcn_s_sleep(2);
      }
      __threadfence();
    }
    __syncthreads();
    for (int k = tid; k < HID; k += 64) hl[k] = hbuf[k];
    __syncthreads();
  }
}

// ---------------- N==1 heads: matvec over y1 (T x 256 f16), fused epilogue --
template<int EPI>
__global__ void k_headvec(const _Float16* __restrict__ Y1,
                          const float* __restrict__ W2, const float* __restrict__ b2,
                          float* __restrict__ outp, int stride, int reps) {
  int t    = blockIdx.x * 8 + (threadIdx.x >> 5);
  int lane = threadIdx.x & 31;
  float acc = 0.f;
  #pragma unroll
  for (int k = lane; k < MLP; k += 32) acc += (float)Y1[(size_t)t * MLP + k] * W2[k];
  acc = waveReduceSum(acc);
  if (lane == 0) {
    float v = acc + b2[0];
    if (EPI == 0)      v = sigmoidf(v) * (DT_MAX - DT_MIN) + DT_MIN;
    else if (EPI == 1) v = fminf(softplusf(v) + G_MIN, G_MAX);
    else               v = U_MAX * tanhf(v);
    for (int r = 0; r < reps; ++r) outp[(size_t)t * stride + r] = v;
  }
}

// ---------------- modified Gram-Schmidt, in place on G region ---------------
__global__ void k_gram_schmidt(float* __restrict__ Gio) {  // T x PHI x RANK
  __shared__ float col[RANK][PHI];   // 32 KB
  __shared__ float red[8];
  __shared__ float bcast;
  const int t   = blockIdx.x;
  const int tid = threadIdx.x;       // 256
  float* base = Gio + (size_t)t * PHI * RANK;
  for (int idx = tid; idx < PHI * RANK; idx += 256) {
    float v = base[idx];
    if (!(v == v)) v = 0.f;          // nan_to_num
    col[idx & (RANK - 1)][idx >> 3] = v;
  }
  __syncthreads();
  for (int k = 0; k < RANK; ++k) {
    for (int q = 0; q < k; ++q) {
      float p = 0.f;
      for (int i = tid; i < PHI; i += 256) p += col[q][i] * col[k][i];
      p = waveReduceSum(p);
      if ((tid & 31) == 0) red[tid >> 5] = p;
      __syncthreads();
      if (tid == 0) { float s = 0.f; for (int w = 0; w < 8; ++w) s += red[w]; bcast = s; }
      __syncthreads();
      float d = bcast;
      for (int i = tid; i < PHI; i += 256) col[k][i] -= d * col[q][i];
      __syncthreads();
    }
    float p = 0.f;
    for (int i = tid; i < PHI; i += 256) { float v = col[k][i]; p += v * v; }
    p = waveReduceSum(p);
    if ((tid & 31) == 0) red[tid >> 5] = p;
    __syncthreads();
    if (tid == 0) { float s = 0.f; for (int w = 0; w < 8; ++w) s += red[w];
                    bcast = fmaxf(sqrtf(s), GS_EPS); }
    __syncthreads();
    float inv = 1.f / bcast;
    for (int i = tid; i < PHI; i += 256) col[k][i] *= inv;
    __syncthreads();
  }
  for (int idx = tid; idx < PHI * RANK; idx += 256)
    base[idx] = col[idx & (RANK - 1)][idx >> 3];
}

// ---------------- host: schedule everything on `stream` ----------------
extern "C" void kernel_launch(void* const* d_in, const int* in_sizes, int n_in,
                              void* d_out, int out_size, void* d_ws, size_t ws_size,
                              hipStream_t stream) {
  (void)in_sizes; (void)n_in; (void)out_size; (void)ws_size;
  const float* feats = (const float*)d_in[0];
  const float* fg    = (const float*)d_in[1];
  const float* fb    = (const float*)d_in[2];
  const float* W_ih  = (const float*)d_in[3];
  const float* W_hh  = (const float*)d_in[4];
  const float* b_ih  = (const float*)d_in[5];
  const float* b_hh  = (const float*)d_in[6];
  // heads: dt=0, gam=1, M=2, G=3, gain=4 ; items: ln_g, ln_b, W1, b1, W2, b2
  auto hp = [&](int head, int item) -> const float* {
    return (const float*)d_in[7 + head * 6 + item];
  };
  float* out = (float*)d_out;
  const size_t o_dt = 0, o_gam = T, o_M = 2 * T;
  const size_t o_G  = o_M + (size_t)T * PHI;
  const size_t o_u  = o_G + (size_t)T * PHI * RANK;

  char* ws = (char*)d_ws;
  size_t off = 0;
  auto alloc = [&](size_t bytes) -> char* {
    char* p = ws + off; off = (off + bytes + 255) & ~(size_t)255; return p;
  };
  _Float16* xh    = (_Float16*)alloc((size_t)T * FEAT * 2);
  _Float16* Wih16 = (_Float16*)alloc((size_t)3 * HID * FEAT * 2);
  float*    gi    = (float*)alloc((size_t)T * 3 * HID * 4);
  float*    hbuf  = (float*)alloc(HID * 4);
  unsigned* syncp = (unsigned*)alloc(256);
  float*    hs    = (float*)alloc((size_t)T * HID * 4);
  _Float16* hln   = (_Float16*)alloc((size_t)5 * T * HID * 2);
  _Float16* W1h   = (_Float16*)alloc((size_t)5 * MLP * HID * 2);
  _Float16* y1    = (_Float16*)alloc((size_t)5 * T * MLP * 2);
  _Float16* W2Mh  = (_Float16*)alloc((size_t)PHI * MLP * 2);
  _Float16* W2Gh  = (_Float16*)alloc((size_t)PHI * RANK * MLP * 2);

  k_init<<<1, 512, 0, stream>>>(hbuf, syncp);

  auto cvt = [&](const float* src, _Float16* dst, int n) {
    k_cvt16<<<(n + 255) / 256, 256, 0, stream>>>(src, dst, n);
  };
  cvt(W_ih, Wih16, 3 * HID * FEAT);
  for (int h = 0; h < 5; ++h) cvt(hp(h, 2), W1h + (size_t)h * MLP * HID, MLP * HID);
  cvt(hp(2, 4), W2Mh, PHI * MLP);
  cvt(hp(3, 4), W2Gh, PHI * RANK * MLP);

  k_feat_ln<<<T / 8, 256, 0, stream>>>(feats, fg, fb, xh);

  // gi = xln @ W_ih^T + b_ih   (M=2048, N=1536, K=256)
  k_gemm16<0><<<dim3(3 * HID / 64, T / 128), 256, 0, stream>>>(
      xh, Wih16, b_ih, gi, nullptr, FEAT, 3 * HID);

  // sequential GRU scan (32 co-resident WGs, LDS weights, global split barrier)
  k_gru_scan<<<SCAN_NWG, 64, 0, stream>>>(gi, W_hh, b_hh, hs, hbuf, syncp);

  // per-head LN of hs -> 5 f16 matrices
  k_head_ln<<<T / 8, 256, 0, stream>>>(hs,
      hp(0, 0), hp(0, 1), hp(1, 0), hp(1, 1), hp(2, 0), hp(2, 1),
      hp(3, 0), hp(3, 1), hp(4, 0), hp(4, 1), hln);

  // y1[h] = GELU(hln[h] @ W1^T + b1)  (M=2048, N=256, K=512), f16 out
  for (int h = 0; h < 5; ++h) {
    k_gemm16<1><<<dim3(MLP / 64, T / 128), 256, 0, stream>>>(
        hln + (size_t)h * T * HID, W1h + (size_t)h * MLP * HID, hp(h, 3),
        nullptr, y1 + (size_t)h * T * MLP, HID, MLP);
  }

  // M head: softplus(y1 @ W2^T + b2) + 0.1 -> d_out  (N=1024)
  k_gemm16<2><<<dim3(PHI / 64, T / 128), 256, 0, stream>>>(
      y1 + (size_t)2 * T * MLP, W2Mh, hp(2, 5), out + o_M, nullptr, MLP, PHI);

  // G head raw: y1 @ W2^T + b2 -> d_out G region  (N=8192)
  k_gemm16<0><<<dim3(PHI * RANK / 64, T / 128), 256, 0, stream>>>(
      y1 + (size_t)3 * T * MLP, W2Gh, hp(3, 5), out + o_G, nullptr, MLP, PHI * RANK);

  // N==1 heads with fused epilogues
  k_headvec<0><<<T / 8, 256, 0, stream>>>(y1,                       hp(0, 4), hp(0, 5), out + o_dt, 1, 1);
  k_headvec<1><<<T / 8, 256, 0, stream>>>(y1 + (size_t)1 * T * MLP, hp(1, 4), hp(1, 5), out + o_gam, 1, 1);
  k_headvec<2><<<T / 8, 256, 0, stream>>>(y1 + (size_t)4 * T * MLP, hp(4, 4), hp(4, 5), out + o_u, CMAX, CMAX);

  // per-timestep modified Gram-Schmidt, in place
  k_gram_schmidt<<<T, 256, 0, stream>>>(out + o_G);
}
